// InteractionBlock_1563368095923
// MI455X (gfx1250) — compile-verified, matrix operand order checked by reading, not run on hardware
//
#include <hip/hip_runtime.h>

// ---------------- problem constants ----------------
#define NNODES 50000
#define NEDGES 800000
#define MULC   8
#define NB     20
#define HIDN   64
#define WN     256      // 4*8*8
#define KP     32       // K of GEMM1 padded 20 -> 32

typedef __attribute__((ext_vector_type(16))) _Float16 v16h;
typedef __attribute__((ext_vector_type(8)))  _Float16 v8h;
typedef __attribute__((ext_vector_type(8)))  float    v8f;

union AFrag { v16h v; v8h h[2]; };

// ---------------- LDS layout for edge kernel ----------------
// static region: f16 weights are one contiguous 36864 B image (W1hT ++ W2hT),
// async-DMA'd from workspace, followed by f32 biases.
constexpr int OFF_W1   = 0;                       // 2048 half  (4096 B)  [64][32] N-major
constexpr int OFF_W2   = 4096;                    // 16384 half (32768 B) [256][64] N-major
constexpr int WBYTES   = 36864;                   // contiguous f16 weight image
constexpr int OFF_B1   = 36864;                   // 64 f32
constexpr int OFF_B2   = 37120;                   // 256 f32
constexpr int OFF_WAVE = 38144;                   // per-wave regions
// per-wave region (6144 B each)
constexpr int PW_RBF = 0;      // [16][32] half (1024 B)
constexpr int PW_HID = 1024;   // [16][64] half (2048 B)
constexpr int PW_HC  = 3072;   // [16][32] f32  (2048 B)
constexpr int PW_U   = 5120;   // [16][8]  f32  (512 B)
constexpr int PW_Y   = 5632;   // [16][4]  f32  (256 B)
constexpr int PW_IDX = 5888;   // row[16], col[16] int (128 B)
constexpr int PW_BYTES = 6144;
constexpr int EDGE_SMEM = OFF_WAVE + 8 * PW_BYTES;   // 87296 B
static_assert(EDGE_SMEM <= 160 * 1024, "LDS budget");

// ---------------- helper kernels ----------------
__global__ void zero_kernel(float* __restrict__ p, int n) {
    int i = blockIdx.x * blockDim.x + threadIdx.x;
    if (i < n) p[i] = 0.f;
}

__global__ void prep_weights(const float* __restrict__ W1, const float* __restrict__ W2,
                             _Float16* __restrict__ w1hT, _Float16* __restrict__ w2hT) {
    int tid = blockIdx.x * blockDim.x + threadIdx.x;
    int stride = gridDim.x * blockDim.x;
    // W1: [20][64] f32 -> [64][32] f16 (K padded with zeros)
    for (int idx = tid; idx < HIDN * KP; idx += stride) {
        int n = idx >> 5, k = idx & 31;
        w1hT[idx] = (k < NB) ? (_Float16)W1[k * HIDN + n] : (_Float16)0.f;
    }
    // W2: [64][256] f32 -> [256][64] f16
    for (int idx = tid; idx < WN * HIDN; idx += stride) {
        int n = idx >> 6, k = idx & 63;
        w2hT[idx] = (_Float16)W2[k * WN + n];
    }
}

// ---------------- edge message kernel (WMMA) ----------------
__launch_bounds__(256)
__global__ void edge_kernel(const float* __restrict__ h, const float* __restrict__ pos,
                            const int* __restrict__ ei, const float* __restrict__ centers,
                            const float* __restrict__ gamma, const float* __restrict__ b1,
                            const float* __restrict__ b2,
                            const _Float16* __restrict__ wHalf,   // w1hT ++ w2hT, 36864 B
                            float* __restrict__ agg) {
    extern __shared__ char smem[];
    _Float16* sW1 = (_Float16*)(smem + OFF_W1);
    _Float16* sW2 = (_Float16*)(smem + OFF_W2);
    float*    sB1 = (float*)(smem + OFF_B1);
    float*    sB2 = (float*)(smem + OFF_B2);

    const int tid  = threadIdx.x;
    const int wave = tid >> 5;
    const int lane = tid & 31;
    char* wbase = smem + OFF_WAVE + wave * PW_BYTES;
    _Float16* sRbf = (_Float16*)(wbase + PW_RBF);
    _Float16* sHid = (_Float16*)(wbase + PW_HID);
    float*    sHc  = (float*)(wbase + PW_HC);
    float*    sU   = (float*)(wbase + PW_U);
    float*    sY   = (float*)(wbase + PW_Y);
    int*      sIdx = (int*)(wbase + PW_IDX);

    // ---- async DMA: stage 36864 B f16 weight image global -> LDS ----
    // CDNA5 GLOBAL_LOAD_ASYNC_TO_LDS (GVS mode), tracked on ASYNCcnt; overlaps
    // with geometry / RBF / feature-gather below.
    {
        const unsigned ldsBase = (unsigned)(uintptr_t)smem;
        #pragma unroll
        for (int c = 0; c < WBYTES / 4096; ++c) {   // 9 chunks of 4096 B / block
            const unsigned off = (unsigned)(tid * 16 + c * 4096);
            asm volatile("global_load_async_to_lds_b128 %0, %1, %2"
                         :: "v"(ldsBase + off), "v"(off), "s"(wHalf)
                         : "memory");
        }
    }
    // biases via normal path (small)
    for (int i = tid; i < HIDN; i += 256) sB1[i] = b1[i];
    for (int i = tid; i < WN; i += 256)   sB2[i] = b2[i];

    // ---- per-edge geometry + RBF (lanes 0..15, one edge each) ----
    if (lane < 16) {
        const int e = blockIdx.x * 128 + wave * 16 + lane;
        const int row = ei[e];
        const int col = ei[NEDGES + e];
        sIdx[lane] = row;
        sIdx[16 + lane] = col;
        const float dx = pos[row * 3 + 0] - pos[col * 3 + 0];
        const float dy = pos[row * 3 + 1] - pos[col * 3 + 1];
        const float dz = pos[row * 3 + 2] - pos[col * 3 + 2];
        const float len = sqrtf(dx * dx + dy * dy + dz * dz);
        const float rinv = 1.7320508075688772f / len;   // sqrt(3)/len
        sY[lane * 4 + 0] = dx * rinv;
        sY[lane * 4 + 1] = dy * rinv;
        sY[lane * 4 + 2] = dz * rinv;
        const float g = gamma[0];
        #pragma unroll
        for (int k = 0; k < KP; ++k) {
            float v = 0.f;
            if (k < NB) {
                const float d = len - centers[k];
                v = __expf(-g * d * d);
            }
            sRbf[lane * KP + k] = (_Float16)v;
        }
    }
    __syncthreads();

    // ---- gather h[col] (32 floats per edge) ----
    for (int i = lane; i < 16 * 32; i += 32) {
        const int e2 = i >> 5, f = i & 31;
        sHc[i] = h[(size_t)sIdx[16 + e2] * 32 + f];
    }
    __syncthreads();

    // ---- u[e][i] = sum_c v_n[e,i,c] * y1[e,c] ----
    for (int i = lane; i < 16 * 8; i += 32) {
        const int e2 = i >> 3, ii = i & 7;
        const float* vp = sHc + e2 * 32 + 8 + ii * 3;
        const float* yp = sY + e2 * 4;
        sU[i] = vp[0] * yp[0] + vp[1] * yp[1] + vp[2] * yp[2];
    }
    // weights must be resident before GEMM1
    asm volatile("s_wait_asynccnt 0x0" ::: "memory");
    __syncthreads();

    // ---- fragment addressing (ISA 16-bit A/B layout) ----
    const int kb   = (lane & 16) ? 8 : 0;   // K base of lane half
    const int m16  = lane & 15;             // row (A) / col (B) within tile
    const int roff = (lane & 16) ? 8 : 0;   // D-row offset of lane half

    // ---- GEMM1: rbf[16x32] x W1[32x64] -> hid (ReLU, f16) ----
    AFrag a;
    a.h[0] = *reinterpret_cast<const v8h*>(sRbf + m16 * KP + kb);
    a.h[1] = *reinterpret_cast<const v8h*>(sRbf + m16 * KP + kb + 16);
    #pragma unroll
    for (int t = 0; t < 4; ++t) {
        const int n = t * 16 + m16;
        AFrag b;
        b.h[0] = *reinterpret_cast<const v8h*>(sW1 + n * KP + kb);
        b.h[1] = *reinterpret_cast<const v8h*>(sW1 + n * KP + kb + 16);
        v8f c = {};
        c = __builtin_amdgcn_wmma_f32_16x16x32_f16(false, a.v, false, b.v, (short)0, c,
                                                   false, false);
        const float bias = sB1[n];
        #pragma unroll
        for (int r = 0; r < 8; ++r) {
            const float hv = fmaxf(c[r] + bias, 0.f);
            sHid[(r + roff) * HIDN + n] = (_Float16)hv;
        }
    }
    __syncthreads();

    // ---- GEMM2: hid[16x64] x W2[64x256], consumed into register accumulators ----
    AFrag a0, a1;
    a0.h[0] = *reinterpret_cast<const v8h*>(sHid + m16 * HIDN + kb);
    a0.h[1] = *reinterpret_cast<const v8h*>(sHid + m16 * HIDN + kb + 16);
    a1.h[0] = *reinterpret_cast<const v8h*>(sHid + m16 * HIDN + 32 + kb);
    a1.h[1] = *reinterpret_cast<const v8h*>(sHid + m16 * HIDN + 32 + kb + 16);

    const float c110 = 0.5773502691896258f;   // 1/sqrt(3)
    float accS[8];        // scalar-channel i-sum (ksel0 + c110*ksel1)
    float accV[8][3];     // vector-channel i-sum (ksel2*y1 + ksel3)
    #pragma unroll
    for (int r = 0; r < 8; ++r) {
        accS[r] = 0.f;
        accV[r][0] = accV[r][1] = accV[r][2] = 0.f;
    }

    #pragma unroll
    for (int ksel = 0; ksel < 4; ++ksel) {
        float accT[8];
        #pragma unroll
        for (int r = 0; r < 8; ++r) accT[r] = 0.f;
        #pragma unroll
        for (int tt = 0; tt < 4; ++tt) {
            const int t = ksel * 4 + tt;
            const int n = t * 16 + m16;
            AFrag b0, b1f;
            b0.h[0]  = *reinterpret_cast<const v8h*>(sW2 + n * HIDN + kb);
            b0.h[1]  = *reinterpret_cast<const v8h*>(sW2 + n * HIDN + kb + 16);
            b1f.h[0] = *reinterpret_cast<const v8h*>(sW2 + n * HIDN + 32 + kb);
            b1f.h[1] = *reinterpret_cast<const v8h*>(sW2 + n * HIDN + 32 + kb + 16);
            v8f acc = {};
            acc = __builtin_amdgcn_wmma_f32_16x16x32_f16(false, a0.v, false, b0.v, (short)0,
                                                         acc, false, false);
            acc = __builtin_amdgcn_wmma_f32_16x16x32_f16(false, a1.v, false, b1f.v, (short)0,
                                                         acc, false, false);
            // tpw column decode: col = ksel*64 + ii*8 + j ; this lane: ii below, j = m16&7
            const int ii = (tt << 1) | (m16 >> 3);
            const float bias = sB2[n];
            #pragma unroll
            for (int r = 0; r < 8; ++r) {
                const int e2 = r + roff;
                const float val = acc[r] + bias;
                if (ksel == 0) {
                    accS[r] += val * sHc[e2 * 32 + ii];
                } else if (ksel == 1) {
                    accS[r] += c110 * val * sU[e2 * 8 + ii];
                } else if (ksel == 2) {
                    accT[r] += val * sHc[e2 * 32 + ii];      // y1 factors out of i-sum
                } else {
                    const float* vp = sHc + e2 * 32 + 8 + ii * 3;
                    accV[r][0] += val * vp[0];
                    accV[r][1] += val * vp[1];
                    accV[r][2] += val * vp[2];
                }
            }
        }
        if (ksel == 2) {
            #pragma unroll
            for (int r = 0; r < 8; ++r) {
                const int e2 = r + roff;
                accV[r][0] += accT[r] * sY[e2 * 4 + 0];
                accV[r][1] += accT[r] * sY[e2 * 4 + 1];
                accV[r][2] += accT[r] * sY[e2 * 4 + 2];
            }
        }
    }

    // ---- combine complementary i-halves (lane ^ 8) and scatter to agg ----
    // alpha = 1/sqrt(2*mul) = 0.25
    const int j = m16 & 7;
    #pragma unroll
    for (int r = 0; r < 8; ++r) {
        const int e2 = r + roff;
        const float s  = accS[r]    + __shfl_xor(accS[r], 8, 32);
        const float v0 = accV[r][0] + __shfl_xor(accV[r][0], 8, 32);
        const float v1 = accV[r][1] + __shfl_xor(accV[r][1], 8, 32);
        const float v2 = accV[r][2] + __shfl_xor(accV[r][2], 8, 32);
        if (m16 < 8) {
            float* ap = agg + (size_t)sIdx[e2] * 32;
            atomicAdd(ap + j, 0.25f * s);
            atomicAdd(ap + 8 + j * 3 + 0, 0.25f * v0);
            atomicAdd(ap + 8 + j * 3 + 1, 0.25f * v1);
            atomicAdd(ap + 8 + j * 3 + 2, 0.25f * v2);
        }
    }
}

// ---------------- node self-interaction + gating ----------------
__global__ void node_kernel(const float* __restrict__ h, const float* __restrict__ Ws,
                            const float* __restrict__ Wv, const float* __restrict__ agg,
                            float* __restrict__ out) {
    __shared__ float sWs[64], sWv[64];
    const int tid = threadIdx.x;
    if (tid < 64) sWs[tid] = Ws[tid];
    else if (tid < 128) sWv[tid - 64] = Wv[tid - 64];
    __syncthreads();

    const int n = blockIdx.x * blockDim.x + tid;
    if (n >= NNODES) return;

    float hn[32];
    const float* hp = h + (size_t)n * 32;
    #pragma unroll
    for (int i = 0; i < 32; ++i) hn[i] = hp[i];
    const float* ap = agg + (size_t)n * 32;
    const float inv = 0.35355339059327373f;   // 1/sqrt(8)

    float ov[32];
    #pragma unroll
    for (int j = 0; j < 8; ++j) {
        float acc = 0.f;
        #pragma unroll
        for (int i = 0; i < 8; ++i) acc += hn[i] * sWs[i * 8 + j];
        const float s = ap[j] + inv * acc;
        const float ns = fabsf(s);
        ov[j] = s * (ns / fmaxf(ns, 1e-12f));
    }
    #pragma unroll
    for (int j = 0; j < 8; ++j) {
        float v[3];
        #pragma unroll
        for (int c = 0; c < 3; ++c) {
            float acc = 0.f;
            #pragma unroll
            for (int i = 0; i < 8; ++i) acc += hn[8 + i * 3 + c] * sWv[i * 8 + j];
            v[c] = ap[8 + j * 3 + c] + inv * acc;
        }
        const float nv = sqrtf(v[0] * v[0] + v[1] * v[1] + v[2] * v[2]);
        const float sc = nv / fmaxf(nv, 1e-12f);
        ov[8 + j * 3 + 0] = v[0] * sc;
        ov[8 + j * 3 + 1] = v[1] * sc;
        ov[8 + j * 3 + 2] = v[2] * sc;
    }
    float* op = out + (size_t)n * 32;
    #pragma unroll
    for (int i = 0; i < 32; ++i) op[i] = ov[i];
}

// ---------------- launcher ----------------
extern "C" void kernel_launch(void* const* d_in, const int* in_sizes, int n_in,
                              void* d_out, int out_size, void* d_ws, size_t ws_size,
                              hipStream_t stream) {
    const float* h       = (const float*)d_in[0];
    const float* pos     = (const float*)d_in[1];
    const int*   ei      = (const int*)d_in[2];
    const float* centers = (const float*)d_in[3];
    const float* gamma   = (const float*)d_in[4];
    const float* W1      = (const float*)d_in[5];
    const float* b1      = (const float*)d_in[6];
    const float* W2      = (const float*)d_in[7];
    const float* b2      = (const float*)d_in[8];
    const float* Ws      = (const float*)d_in[9];
    const float* Wv      = (const float*)d_in[10];
    float*       out     = (float*)d_out;

    // workspace layout: agg (6.4 MB) | contiguous f16 weight image: W1 (4 KB) ++ W2 (32 KB)
    constexpr size_t AGG_BYTES = (size_t)NNODES * 32 * sizeof(float);
    float*    agg  = (float*)d_ws;
    _Float16* w1hT = (_Float16*)((char*)d_ws + AGG_BYTES);
    _Float16* w2hT = (_Float16*)((char*)d_ws + AGG_BYTES + 4096);

    const int aggN = NNODES * 32;
    zero_kernel<<<(aggN + 255) / 256, 256, 0, stream>>>(agg, aggN);
    prep_weights<<<32, 256, 0, stream>>>(W1, W2, w1hT, w2hT);

    // 8 waves/block * 16 edges/wave = 128 edges/block; 800000/128 = 6250 exactly
    edge_kernel<<<NEDGES / 128, 256, EDGE_SMEM, stream>>>(h, pos, ei, centers, gamma, b1, b2,
                                                          w1hT, agg);
    node_kernel<<<(NNODES + 255) / 256, 256, 0, stream>>>(h, Ws, Wv, agg, out);
}
